// BahdanauAttentionLSTMDecoder_61529701482994
// MI455X (gfx1250) — compile-verified, matrix operand order checked by reading, not run on hardware
//
#include <hip/hip_runtime.h>
#include <hip/hip_bf16.h>

// ---------------------------------------------------------------------------
// Problem constants (match the JAX reference)
// ---------------------------------------------------------------------------
#define BB 64
#define SS 256
#define TT 256
#define UU 1024
#define DD 1024
#define EE 1024
#define VV 1024

typedef __attribute__((ext_vector_type(16))) __bf16 v16bf;
typedef __attribute__((ext_vector_type(8)))  __bf16 v8bf;
typedef __attribute__((ext_vector_type(8)))  float  v8f;

#define WAVES_PER_BLOCK 4
#define GEMM_BLOCK (WAVES_PER_BLOCK * 32)

// ---------------------------------------------------------------------------
// Pure-bf16 WMMA GEMM, M-blocked by 4 tiles:
//   out[M,N](+bias) = A[M,K](bf16) x W(bf16 column-major: Wt[n*K+k]).
// One wave owns a 64-row x 16-col strip.  The K loop is split into (up to 3)
// concat segments; inside a segment all addresses are pure pointer bumps
// (+32 elements), so each iteration is 10 independent b128 loads + 4
// back-to-back v_wmma_f32_16x16x32_bf16 sharing one B fragment (reuse_b hint).
// ---------------------------------------------------------------------------
__global__ __launch_bounds__(GEMM_BLOCK)
void gemm_wmma(const __bf16* __restrict__ Wt, int K,
               const __bf16* __restrict__ A0, int lda0, int k1,
               const __bf16* __restrict__ A1, int lda1, int k2,
               const __bf16* __restrict__ A2, int lda2,
               const float* __restrict__ bias,
               float* __restrict__ outf, __bf16* __restrict__ outb, int ldc,
               int M, int N)
{
    const int ntn   = N >> 4;
    const int total = (M >> 6) * ntn;          // M is a multiple of 64
    const int wid   = blockIdx.x * WAVES_PER_BLOCK + ((int)threadIdx.x >> 5);
    if (wid >= total) return;                  // whole-wave uniform exit

    const int mg   = wid / ntn;
    const int nt   = wid - mg * ntn;
    const int lane = threadIdx.x & 31;
    const int l    = lane & 15;                // row (A) / col (B) within tile
    const int hi   = lane >> 4;                // K sub-chunk selector (ISA 16-bit layout)

    const int n = (nt << 4) + l;
    const __bf16* __restrict__ brow = Wt + (size_t)n * (size_t)K + (hi << 3);
    const int mrow0 = (mg << 6) + l;

    v8f acc0 = {}, acc1 = {}, acc2 = {}, acc3 = {};

    auto mk16 = [](v8bf lo, v8bf hh) -> v16bf {
        v16bf r;
#pragma unroll
        for (int i = 0; i < 8; ++i) { r[i] = lo[i]; r[i + 8] = hh[i]; }
        return r;
    };

    // per-segment inner loop: pointer-bump addressing only
    auto run_seg = [&](const __bf16* __restrict__ base, int ld, int kbeg, int kend) {
        if (kbeg >= kend) return;
        const __bf16* ap0 = base + (size_t)(mrow0 +  0) * (size_t)ld + (hi << 3);
        const __bf16* ap1 = base + (size_t)(mrow0 + 16) * (size_t)ld + (hi << 3);
        const __bf16* ap2 = base + (size_t)(mrow0 + 32) * (size_t)ld + (hi << 3);
        const __bf16* ap3 = base + (size_t)(mrow0 + 48) * (size_t)ld + (hi << 3);
        const __bf16* bp  = brow + kbeg;
        for (int k0 = kbeg; k0 < kend; k0 += 32) {
            // 10 independent 128-bit loads -> one clause, one wait
            v8bf b0  = *((const v8bf*)bp);
            v8bf b1  = *((const v8bf*)(bp + 16));
            v8bf x00 = *((const v8bf*)ap0), x01 = *((const v8bf*)(ap0 + 16));
            v8bf x10 = *((const v8bf*)ap1), x11 = *((const v8bf*)(ap1 + 16));
            v8bf x20 = *((const v8bf*)ap2), x21 = *((const v8bf*)(ap2 + 16));
            v8bf x30 = *((const v8bf*)ap3), x31 = *((const v8bf*)(ap3 + 16));
            bp  += 32;
            ap0 += 32; ap1 += 32; ap2 += 32; ap3 += 32;

            v16bf bfrag = mk16(b0, b1);
            v16bf a0 = mk16(x00, x01);
            v16bf a1 = mk16(x10, x11);
            v16bf a2 = mk16(x20, x21);
            v16bf a3 = mk16(x30, x31);
            // 4 independent accumulators, same B regs + same opcode back-to-back
            // => reuse_b operand-cache hint legal on ops 2..4 (ISA 7.12).
            acc0 = __builtin_amdgcn_wmma_f32_16x16x32_bf16(false, a0, false, bfrag, (short)0, acc0, false, false);
            acc1 = __builtin_amdgcn_wmma_f32_16x16x32_bf16(false, a1, false, bfrag, (short)0, acc1, false, true);
            acc2 = __builtin_amdgcn_wmma_f32_16x16x32_bf16(false, a2, false, bfrag, (short)0, acc2, false, true);
            acc3 = __builtin_amdgcn_wmma_f32_16x16x32_bf16(false, a3, false, bfrag, (short)0, acc3, false, true);
        }
    };

    run_seg(A0, lda0, 0,  k1);
    run_seg(A1, lda1, k1, k2);
    run_seg(A2, lda2, k2, K);

    // --- store: lanes<16 -> rows +0..7, lanes>=16 -> rows +8..15; col = nt*16+l
    const int col  = (nt << 4) + l;
    const float bv = bias ? bias[col] : 0.0f;
    auto store1 = [&](const v8f& acc, int mi) {
        const int row0 = (mg << 6) + (mi << 4) + (hi << 3);
        if (outf) {
#pragma unroll
            for (int r = 0; r < 8; ++r)
                outf[(size_t)(row0 + r) * (size_t)ldc + col] = acc[r] + bv;
        } else {
#pragma unroll
            for (int r = 0; r < 8; ++r)
                outb[(size_t)(row0 + r) * (size_t)ldc + col] = (__bf16)(acc[r] + bv);
        }
    };
    store1(acc0, 0); store1(acc1, 1); store1(acc2, 2); store1(acc3, 3);
}

// ---------------------------------------------------------------------------
// Weight convert + transpose: W[K][N] fp32 row-major -> Wt[n*ldt + koff + k] bf16
// ---------------------------------------------------------------------------
__global__ __launch_bounds__(256)
void convert_transpose(const float* __restrict__ W, __bf16* __restrict__ Wt,
                       int K, int N, int ldt, int koff)
{
    size_t idx   = (size_t)blockIdx.x * 256 + threadIdx.x;
    size_t total = (size_t)K * (size_t)N;
    if (idx >= total) return;
    int k = (int)(idx / (size_t)N);
    int n = (int)(idx - (size_t)k * (size_t)N);     // coalesced read
    Wt[(size_t)n * (size_t)ldt + (size_t)(koff + k)] = (__bf16)W[idx];
}

// ---------------------------------------------------------------------------
// fp32 -> bf16 flat pack (dec_inputs / memory, once per launch)
// ---------------------------------------------------------------------------
__global__ __launch_bounds__(256)
void pack_bf16(const float* __restrict__ src, __bf16* __restrict__ dst, size_t n)
{
    size_t idx = (size_t)blockIdx.x * 256 + threadIdx.x;
    if (idx < n) dst[idx] = (__bf16)src[idx];
}

// h0 fp32 [B,U] -> bf16 into Ahc[:, 0:U] (row stride 2U)
__global__ __launch_bounds__(256)
void pack_h0(const float* __restrict__ h0, __bf16* __restrict__ Ahc)
{
    int idx = blockIdx.x * 256 + threadIdx.x;
    if (idx >= BB * UU) return;
    int b = idx >> 10, u = idx & (UU - 1);
    Ahc[(size_t)b * (2 * UU) + u] = (__bf16)h0[idx];
}

// ---------------------------------------------------------------------------
// LSTM gate fusion: z[B,4U] (i,f,g,o), c fp32 in-place, h -> bf16 Ahc[:,0:U]
// ---------------------------------------------------------------------------
__global__ __launch_bounds__(256)
void gates_kernel(const float* __restrict__ z, float* __restrict__ c,
                  __bf16* __restrict__ Ahc)
{
    int idx = blockIdx.x * 256 + threadIdx.x;
    if (idx >= BB * UU) return;
    int b = idx >> 10, u = idx & (UU - 1);
    const float* zr = z + (size_t)b * (4 * UU);
    float zi = zr[u], zf = zr[UU + u], zg = zr[2 * UU + u], zo = zr[3 * UU + u];
    float ig = 1.0f / (1.0f + __expf(-zi));
    float fg = 1.0f / (1.0f + __expf(-zf));
    float og = 1.0f / (1.0f + __expf(-zo));
    float cn = fg * c[idx] + ig * tanhf(zg);
    c[idx] = cn;
    Ahc[(size_t)b * (2 * UU) + u] = (__bf16)(og * tanhf(cn));
}

// ---------------------------------------------------------------------------
// Bahdanau attention: score = v . tanh(keys + q), masked softmax over S,
// context -> bf16 into Ahc[:, U:2U].  One workgroup (8 waves) per batch.
// keys(bf16) + membf(bf16) + weights all stay resident in the 192MB L2.
// ---------------------------------------------------------------------------
__global__ __launch_bounds__(256)
void attention_kernel(const __bf16* __restrict__ keys, const float* __restrict__ q,
                      const float* __restrict__ v_att, const __bf16* __restrict__ membf,
                      const int* __restrict__ mem_len, __bf16* __restrict__ Ahc)
{
    const int b = blockIdx.x;
    const int tid = threadIdx.x, wave = tid >> 5, lane = tid & 31;
    __shared__ float s_score[SS];
    __shared__ float s_red[8];
    __shared__ float s_tot;

    const int len = mem_len[b];
    const __bf16* kb = keys + (size_t)b * SS * UU;
    const float*  qb = q + (size_t)b * UU;

    // phase 1: additive scores (wave per s-row, lane-strided dot over U)
    for (int s = wave; s < SS; s += 8) {
        const __bf16* kr = kb + (size_t)s * UU;
        float acc = 0.0f;
#pragma unroll 4
        for (int u = lane; u < UU; u += 32)
            acc = __builtin_fmaf(v_att[u], tanhf((float)kr[u] + qb[u]), acc);
#pragma unroll
        for (int off = 16; off; off >>= 1) acc += __shfl_xor(acc, off);
        if (lane == 0) s_score[s] = (s < len) ? acc : -1e9f;
    }
    __syncthreads();

    // phase 2: masked softmax over S=256 (one score per thread)
    float sc = s_score[tid];
    float mx = sc;
#pragma unroll
    for (int off = 16; off; off >>= 1) mx = fmaxf(mx, __shfl_xor(mx, off));
    if (lane == 0) s_red[wave] = mx;
    __syncthreads();
    if (tid == 0) {
        float v = s_red[0];
        for (int i = 1; i < 8; ++i) v = fmaxf(v, s_red[i]);
        s_tot = v;
    }
    __syncthreads();
    float e = __expf(sc - s_tot);
    float sum = e;
#pragma unroll
    for (int off = 16; off; off >>= 1) sum += __shfl_xor(sum, off);
    __syncthreads();                    // order s_red reads above vs writes below
    if (lane == 0) s_red[wave] = sum;
    __syncthreads();
    if (tid == 0) {
        float v = 0.0f;
        for (int i = 0; i < 8; ++i) v += s_red[i];
        s_tot = 1.0f / v;
    }
    __syncthreads();
    s_score[tid] = e * s_tot;           // normalized alignment
    __syncthreads();

    // phase 3: context[b,d] = sum_s align[s]*memory[b,s,d] -> bf16 Ahc[:,U:2U]
    const __bf16* mb = membf + (size_t)b * SS * DD;
    for (int d = tid; d < DD; d += 256) {
        float acc = 0.0f;
#pragma unroll 4
        for (int s = 0; s < SS; ++s)
            acc = __builtin_fmaf(s_score[s], (float)mb[(size_t)s * DD + d], acc);
        Ahc[(size_t)b * (2 * UU) + UU + d] = (__bf16)acc;
    }
}

// ---------------------------------------------------------------------------
// Host-side orchestration (graph-capture safe)
// ---------------------------------------------------------------------------
static inline size_t align256(size_t x) { return (x + 255) & ~(size_t)255; }
static inline int gemm_blocks(int M, int N) {
    return ((M >> 6) * (N >> 4) + WAVES_PER_BLOCK - 1) / WAVES_PER_BLOCK;
}

extern "C" void kernel_launch(void* const* d_in, const int* in_sizes, int n_in,
                              void* d_out, int out_size, void* d_ws, size_t ws_size,
                              hipStream_t stream)
{
    const float* memory   = (const float*)d_in[0];   // [B,S,D]
    const float* dec_in   = (const float*)d_in[1];   // [B,T,E]
    const float* h0       = (const float*)d_in[2];   // [B,U]
    const float* c0       = (const float*)d_in[3];   // [B,U]
    const float* W_lstm   = (const float*)d_in[4];   // [E+U, 4U]
    const float* U_lstm   = (const float*)d_in[5];   // [U, 4U]
    const float* b_lstm   = (const float*)d_in[6];   // [4U]
    const float* W_mem    = (const float*)d_in[7];   // [D, U]
    const float* W_query  = (const float*)d_in[8];   // [U, U]
    const float* v_att    = (const float*)d_in[9];   // [U]
    const float* W_attn   = (const float*)d_in[10];  // [U+D, U]
    const float* W_out    = (const float*)d_in[11];  // [U, V]
    const float* b_out    = (const float*)d_in[12];  // [V]
    const int*   mem_len  = (const int*)d_in[13];    // [B]
    float* out = (float*)d_out;                      // [B,T,V]

    // ---- workspace carve-up (~138 MB)
    char* ws = (char*)d_ws;
    size_t off = 0;
    __bf16* Wz     = (__bf16*)(ws + off); off = align256(off + (size_t)4096 * 3072 * 2);
    __bf16* Wq     = (__bf16*)(ws + off); off = align256(off + (size_t)1024 * 1024 * 2);
    __bf16* Wa     = (__bf16*)(ws + off); off = align256(off + (size_t)1024 * 2048 * 2);
    __bf16* Wo     = (__bf16*)(ws + off); off = align256(off + (size_t)1024 * 1024 * 2);
    __bf16* Wm     = (__bf16*)(ws + off); off = align256(off + (size_t)1024 * 1024 * 2);
    __bf16* decbf  = (__bf16*)(ws + off); off = align256(off + (size_t)BB * TT * EE * 2);
    __bf16* membf  = (__bf16*)(ws + off); off = align256(off + (size_t)BB * SS * DD * 2);
    __bf16* keysbf = (__bf16*)(ws + off); off = align256(off + (size_t)BB * SS * UU * 2);
    __bf16* Ahc    = (__bf16*)(ws + off); off = align256(off + (size_t)BB * 2 * UU * 2); // [h | ctx]
    __bf16* attnbf = (__bf16*)(ws + off); off = align256(off + (size_t)BB * UU * 2);
    float*  z      = (float*)(ws + off);  off = align256(off + (size_t)BB * 4 * UU * 4);
    float*  c      = (float*)(ws + off);  off = align256(off + (size_t)BB * UU * 4);
    float*  q      = (float*)(ws + off);  off = align256(off + (size_t)BB * UU * 4);
    (void)ws_size; (void)in_sizes; (void)n_in; (void)out_size;

    auto cvgrid = [](size_t n) { return (unsigned)((n + 255) / 256); };

    // ---- per-call weight convert + transpose to bf16 column-major
    convert_transpose<<<cvgrid((size_t)2048 * 4096), 256, 0, stream>>>(W_lstm,  Wz, 2048, 4096, 3072, 0);
    convert_transpose<<<cvgrid((size_t)1024 * 4096), 256, 0, stream>>>(U_lstm,  Wz, 1024, 4096, 3072, 2048);
    convert_transpose<<<cvgrid((size_t)1024 * 1024), 256, 0, stream>>>(W_query, Wq, 1024, 1024, 1024, 0);
    convert_transpose<<<cvgrid((size_t)2048 * 1024), 256, 0, stream>>>(W_attn,  Wa, 2048, 1024, 2048, 0);
    convert_transpose<<<cvgrid((size_t)1024 * 1024), 256, 0, stream>>>(W_out,   Wo, 1024, 1024, 1024, 0);
    convert_transpose<<<cvgrid((size_t)1024 * 1024), 256, 0, stream>>>(W_mem,   Wm, 1024, 1024, 1024, 0);

    // ---- activations to bf16 (A-side of all GEMMs + attention memory reads)
    pack_bf16<<<cvgrid((size_t)BB * TT * EE), 256, 0, stream>>>(dec_in, decbf, (size_t)BB * TT * EE);
    pack_bf16<<<cvgrid((size_t)BB * SS * DD), 256, 0, stream>>>(memory, membf, (size_t)BB * SS * DD);

    // ---- keys = memory @ W_mem  (M=16384, N=1024, K=1024, bf16 output)
    gemm_wmma<<<gemm_blocks(BB * SS, UU), GEMM_BLOCK, 0, stream>>>(
        Wm, DD, membf, DD, DD, nullptr, 0, DD, nullptr, 0,
        nullptr, nullptr, keysbf, UU, BB * SS, UU);

    // ---- recurrent state init
    pack_h0<<<cvgrid((size_t)BB * UU), 256, 0, stream>>>(h0, Ahc);
    (void)hipMemcpyAsync(c, c0, (size_t)BB * UU * 4, hipMemcpyDeviceToDevice, stream);
    (void)hipMemsetAsync(attnbf, 0, (size_t)BB * UU * 2, stream);

    const int blkZ = gemm_blocks(BB, 4 * UU);   // 64 blocks
    const int blkS = gemm_blocks(BB, UU);       // 16 blocks

    for (int t = 0; t < TT; ++t) {
        // z = [x_t | attn | h] @ Wz + b_lstm   (K=3072, N=4096)
        gemm_wmma<<<blkZ, GEMM_BLOCK, 0, stream>>>(
            Wz, 3072,
            decbf + (size_t)t * EE, TT * EE, EE,    // seg0: x_t (row stride T*E)
            attnbf, UU, 2 * UU,                     // seg1: prev attention
            Ahc, 2 * UU,                            // seg2: prev h (Ahc[:,0:U])
            b_lstm, z, nullptr, 4 * UU, BB, 4 * UU);

        // LSTM gates -> c (fp32), h_new (bf16 into Ahc[:,0:U])
        gates_kernel<<<(BB * UU + 255) / 256, 256, 0, stream>>>(z, c, Ahc);

        // q = h_new @ W_query
        gemm_wmma<<<blkS, GEMM_BLOCK, 0, stream>>>(
            Wq, UU, Ahc, 2 * UU, UU, nullptr, 0, UU, nullptr, 0,
            nullptr, q, nullptr, UU, BB, UU);

        // Bahdanau score + softmax + context (ctx -> bf16 Ahc[:,U:2U])
        attention_kernel<<<BB, 256, 0, stream>>>(keysbf, q, v_att, membf, mem_len, Ahc);

        // attn = [h_new | context] @ W_attn  (single segment, K=2048) -> bf16
        gemm_wmma<<<blkS, GEMM_BLOCK, 0, stream>>>(
            Wa, 2 * UU, Ahc, 2 * UU, 2 * UU, nullptr, 0, 2 * UU, nullptr, 0,
            nullptr, nullptr, attnbf, UU, BB, UU);

        // out_t = attn @ W_out + b_out -> d_out[:, t, :]  (row stride T*V)
        gemm_wmma<<<blkS, GEMM_BLOCK, 0, stream>>>(
            Wo, UU, attnbf, UU, UU, nullptr, 0, UU, nullptr, 0,
            b_out, out + (size_t)t * VV, nullptr, TT * VV, BB, VV);
    }
}
